// BiLSTM_CRF_26886495272966
// MI455X (gfx1250) — compile-verified, matrix operand order checked by reading, not run on hardware
//
#include <hip/hip_runtime.h>
#include <hip/hip_bf16.h>

// ---------------------------------------------------------------------------
// BiLSTM-CRF for MI455X (gfx1250): f16 WMMA for all GEMMs, recurrences fused
// into persistent blocks. v4: branch-free fast sigmoid/tanh (v_exp_f32 +
// v_rcp_f32) in the recurrent gate math — removes libm tanhf's exec-mask
// divergence from the timestep critical path. One (m-tile, hidden-tile)
// group per wave everywhere (no spills); k-loops not unrolled.
// B=32 T=128 LC=16 EC=64 hc=64 hw=256 H=512 K=12 START=10 STOP=11
// ---------------------------------------------------------------------------

typedef __attribute__((ext_vector_type(16))) _Float16 v16h;
typedef __attribute__((ext_vector_type(8)))  float    v8f;

__device__ __forceinline__ v8f wmma16(v16h a, v16h b, v8f c) {
    return __builtin_amdgcn_wmma_f32_16x16x32_f16(
        /*neg_a=*/false, a, /*neg_b=*/false, b,
        /*c_mod=*/(short)0, c, /*reuse_a=*/false, /*reuse_b=*/false);
}

// Fragment load for 16x16x32 f16 WMMA from a row-major f16 matrix (ld in
// elements). Works for both A (rows = M) and B-as-W[N][K] (rows = N) since
// we compute A @ W^T. Per ISA layout: lanes 0-15 hold K {0..7,16..23},
// lanes 16-31 hold K {8..15,24..31} of row (lane&15).
__device__ __forceinline__ v16h ldfrag(const _Float16* base, int ld,
                                       int rBase, int kBase, int lane) {
    int r  = rBase + (lane & 15);
    int k0 = kBase + ((lane >> 4) << 3);
    const _Float16* p = base + (size_t)r * ld + k0;
    v16h v;
#pragma unroll
    for (int q = 0; q < 8; ++q) { v[q] = p[q]; v[8 + q] = p[16 + q]; }
    return v;
}

// branch-free transcendentals: v_exp_f32 + v_rcp_f32, no exec-mask paths
__device__ __forceinline__ float sigm(float x) {
    return __builtin_amdgcn_rcpf(1.f + __expf(-x));
}
__device__ __forceinline__ float ftanh(float x) {
    float xc = fminf(fmaxf(x, -15.f), 15.f);          // avoid inf*0 at |x| big
    float e  = __expf(2.f * xc);
    return (e - 1.f) * __builtin_amdgcn_rcpf(e + 1.f);
}

// gather one timestep of char embeddings into LDS tile (32 seq x 64);
// 8 threads per sequence, one uint4 (8 halves) each.
__device__ __forceinline__ void gather32(const int* __restrict__ chars, int seq0,
                                         int t, const _Float16* sCE,
                                         _Float16* dst, int tid) {
    int s = tid >> 3, hsel = (tid & 7) * 8;
    int idx = chars[(seq0 + s) * 16 + t];
    *(uint4*)(dst + s * 64 + hsel) = *(const uint4*)(sCE + idx * 64 + hsel);
}

// ------------------------------- prep --------------------------------------
__global__ void cvt_f32_f16(const float* __restrict__ src,
                            _Float16* __restrict__ dst, int n) {
    int i = blockIdx.x * blockDim.x + threadIdx.x;
    if (i < n) dst[i] = (_Float16)src[i];
}

// W2t (12x512) -> padded 16x512 f16, rows 12..15 = 0
__global__ void pad_w2t(const float* __restrict__ src, _Float16* __restrict__ dst) {
    int i = blockIdx.x * blockDim.x + threadIdx.x;   // 16*512
    if (i >= 16 * 512) return;
    int r = i >> 9, c = i & 511;
    dst[i] = (r < 12) ? (_Float16)src[r * 512 + c] : (_Float16)0.f;
}

// ------------------------- char forward LSTM -------------------------------
// 128 blocks x 256 threads; each block owns 32 sequences, runs 16 steps.
// 8 waves = 2 m-tiles x 4 hidden-col tiles; each wave owns ONE group:
// 4 gate accumulators + cell state in VGPRs; ping-pong sX/sH; 1 barrier/step.
__global__ void __launch_bounds__(256)
char_fwd_kernel(const int* __restrict__ chars, const _Float16* __restrict__ CE,
                const _Float16* __restrict__ Wih, const _Float16* __restrict__ Whh,
                const float* __restrict__ bias, _Float16* __restrict__ xw) {
    __shared__ _Float16 sCE[128 * 64];
    __shared__ _Float16 sX[2][32 * 64];
    __shared__ _Float16 sH[2][32 * 64];

    const int tid  = threadIdx.x;
    const int lane = tid & 31;
    const int wave = tid >> 5;
    const int seq0 = blockIdx.x * 32;

    {   // char-emb table -> LDS (16KB)
        const uint4* s = (const uint4*)CE;
        uint4* d = (uint4*)sCE;
        for (int i = tid; i < 1024; i += 256) d[i] = s[i];
    }
    for (int i = tid; i < 2048; i += 256) sH[0][i] = (_Float16)0.f;
    __syncthreads();                       // sCE visible for gather
    gather32(chars, seq0, 0, sCE, sX[0], tid);

    const int mT  = (wave & 1) << 4;       // m-tile 0..1
    const int hT  = (wave >> 1) << 4;      // hidden-col tile 0..3
    const int hid = hT + (lane & 15);
    const int m0  = mT + ((lane >> 4) << 3);
    float gbias[4];
#pragma unroll
    for (int gate = 0; gate < 4; ++gate) gbias[gate] = bias[gate * 64 + hid];
    float creg[8];
#pragma unroll
    for (int j = 0; j < 8; ++j) creg[j] = 0.f;
    __syncthreads();

    int cur = 0;
    for (int t = 0; t < 16; ++t) {
        const int nxt = cur ^ 1;
        v8f acc[4];
#pragma unroll
        for (int gate = 0; gate < 4; ++gate)
#pragma unroll
            for (int j = 0; j < 8; ++j) acc[gate][j] = gbias[gate];
#pragma unroll 1
        for (int kk = 0; kk < 64; kk += 32) {
            v16h aX = ldfrag(sX[cur], 64, mT, kk, lane);
            v16h aH = ldfrag(sH[cur], 64, mT, kk, lane);
#pragma unroll
            for (int gate = 0; gate < 4; ++gate) {
                int nB = gate * 64 + hT;
                acc[gate] = wmma16(aX, ldfrag(Wih, 64, nB, kk, lane), acc[gate]);
                acc[gate] = wmma16(aH, ldfrag(Whh, 64, nB, kk, lane), acc[gate]);
            }
        }
#pragma unroll
        for (int j = 0; j < 8; ++j) {
            float c = sigm(acc[1][j]) * creg[j] + sigm(acc[0][j]) * ftanh(acc[2][j]);
            float h = sigm(acc[3][j]) * ftanh(c);
            creg[j] = c;
            if (t == 15)
                xw[(size_t)(seq0 + m0 + j) * 384 + hid] = (_Float16)h;
            else
                sH[nxt][(m0 + j) * 64 + hid] = (_Float16)h;
        }
        if (t < 15) gather32(chars, seq0, t + 1, sCE, sX[nxt], tid);
        __syncthreads();
        cur = nxt;
    }
}

// -------------------- char backward = ONE LSTM step ------------------------
// hs_bwd[:, -1] == one step on x[:, LC-1] from zero state; one group per
// wave, gates in registers, result written straight to xw cols 64..127.
__global__ void __launch_bounds__(256)
char_bwd_kernel(const int* __restrict__ chars, const _Float16* __restrict__ CE,
                const _Float16* __restrict__ Wih, const float* __restrict__ bias,
                _Float16* __restrict__ xw) {
    __shared__ _Float16 sCE[128 * 64];
    __shared__ _Float16 sX [32 * 64];
    const int tid = threadIdx.x, lane = tid & 31, wave = tid >> 5;
    const int seq0 = blockIdx.x * 32;
    {
        const uint4* s = (const uint4*)CE;
        uint4* d = (uint4*)sCE;
        for (int i = tid; i < 1024; i += 256) d[i] = s[i];
    }
    __syncthreads();
    gather32(chars, seq0, 15, sCE, sX, tid);
    __syncthreads();

    const int mT  = (wave & 1) << 4;
    const int hT  = (wave >> 1) << 4;
    const int hid = hT + (lane & 15);
    const int m0  = mT + ((lane >> 4) << 3);
    v8f acc[4];
#pragma unroll
    for (int gate = 0; gate < 4; ++gate) {
        float bn = bias[gate * 64 + hid];
#pragma unroll
        for (int j = 0; j < 8; ++j) acc[gate][j] = bn;
    }
#pragma unroll 1
    for (int kk = 0; kk < 64; kk += 32) {
        v16h aX = ldfrag(sX, 64, mT, kk, lane);
#pragma unroll
        for (int gate = 0; gate < 4; ++gate)
            acc[gate] = wmma16(aX, ldfrag(Wih, 64, gate * 64 + hT, kk, lane), acc[gate]);
    }
#pragma unroll
    for (int j = 0; j < 8; ++j) {
        float c = sigm(acc[0][j]) * ftanh(acc[2][j]);   // f-gate * 0 drops
        float h = sigm(acc[3][j]) * ftanh(c);
        xw[(size_t)(seq0 + m0 + j) * 384 + 64 + hid] = (_Float16)h;
    }
}

// ------------------------- word embedding gather ---------------------------
__global__ void gather_we_kernel(const int* __restrict__ sentence,
                                 const float* __restrict__ wemb,
                                 _Float16* __restrict__ xw) {
    int i = blockIdx.x * blockDim.x + threadIdx.x;  // 4096*256
    int s = i >> 8, e = i & 255;
    xw[(size_t)s * 384 + 128 + e] = (_Float16)wemb[(size_t)sentence[s] * 256 + e];
}

// -------------- word-LSTM input GEMM (hoisted out of recurrence) -----------
// Xw[4096x1024] = xw[4096x384] @ Wih^T  (per direction, blockIdx.z)
__global__ void __launch_bounds__(256)
word_in_gemm(const _Float16* __restrict__ xw,
             const _Float16* __restrict__ WihF, const _Float16* __restrict__ WihB,
             _Float16* __restrict__ XF, _Float16* __restrict__ XB) {
    const _Float16* W = blockIdx.z ? WihB : WihF;
    _Float16* X       = blockIdx.z ? XB   : XF;
    const int lane = threadIdx.x & 31, wave = threadIdx.x >> 5;
    const int mBlk = blockIdx.x * 128;
    const int nT   = blockIdx.y * 128 + (wave << 4);
    v8f acc[8];
#pragma unroll
    for (int m = 0; m < 8; ++m)
#pragma unroll
        for (int j = 0; j < 8; ++j) acc[m][j] = 0.f;
#pragma unroll 1
    for (int k = 0; k < 384; k += 32) {
        v16h b = ldfrag(W, 384, nT, k, lane);        // reused across 8 M tiles
#pragma unroll
        for (int m = 0; m < 8; ++m)
            acc[m] = wmma16(ldfrag(xw, 384, mBlk + (m << 4), k, lane), b, acc[m]);
    }
    int n = nT + (lane & 15);
#pragma unroll
    for (int m = 0; m < 8; ++m) {
        int m0 = mBlk + (m << 4) + ((lane >> 4) << 3);
#pragma unroll
        for (int j = 0; j < 8; ++j)
            X[(size_t)(m0 + j) * 1024 + n] = (_Float16)acc[m][j];
    }
}

// --------------------- word recurrent LSTM (persistent) --------------------
// 4 blocks = (direction x batch-half), 512 threads = 16 waves; each wave owns
// ONE hidden-col tile (all 4 gates) for 16 batch rows; cell state in VGPRs;
// only h goes through LDS (ping-pong), one barrier per step.
__global__ void __launch_bounds__(512)
word_rec_kernel(const _Float16* __restrict__ XF, const _Float16* __restrict__ XB,
                const _Float16* __restrict__ WhhF, const _Float16* __restrict__ WhhB,
                const float* __restrict__ biasF, const float* __restrict__ biasB,
                _Float16* __restrict__ hAll) {
    const int dir = blockIdx.x & 1;
    const int bh  = blockIdx.x >> 1;      // batch half: rows bh*16..bh*16+15
    const _Float16* X   = dir ? XB   : XF;
    const _Float16* Whh = dir ? WhhB : WhhF;
    const float* bias   = dir ? biasB : biasF;
    __shared__ _Float16 sH[2][16 * 256];
    const int tid = threadIdx.x, lane = tid & 31, wave = tid >> 5;

    // warm L2 with the full Whh (512KB) before the sequential loop
    for (size_t p = (size_t)tid * 256; p < (size_t)512 * 1024; p += (size_t)512 * 256)
        __builtin_prefetch((const char*)Whh + p, 0, 1);

    for (int i = tid; i < 4096; i += 512) sH[0][i] = (_Float16)0.f;

    const int hT  = wave << 4;            // hidden-col tile 0..15
    const int hid = hT + (lane & 15);
    const int m0  = (lane >> 4) << 3;     // local batch row base
    const int bg0 = bh * 16 + m0;         // global batch row base
    float gbias[4];
#pragma unroll
    for (int gate = 0; gate < 4; ++gate) gbias[gate] = bias[gate * 256 + hid];
    float creg[8];
#pragma unroll
    for (int j = 0; j < 8; ++j) creg[j] = 0.f;
    __syncthreads();

    int cur = 0;
    for (int st = 0; st < 128; ++st) {
        const int t = dir ? 127 - st : st;
        const int nxt = cur ^ 1;
        v8f acc[4];
#pragma unroll
        for (int gate = 0; gate < 4; ++gate) {
            const int n = gate * 256 + hid;
#pragma unroll
            for (int j = 0; j < 8; ++j)
                acc[gate][j] = gbias[gate] +
                    (float)X[(size_t)((bg0 + j) * 128 + t) * 1024 + n];
        }
#pragma unroll 1
        for (int k = 0; k < 256; k += 32) {
            v16h aH = ldfrag(sH[cur], 256, 0, k, lane);
#pragma unroll
            for (int gate = 0; gate < 4; ++gate)
                acc[gate] = wmma16(aH, ldfrag(Whh, 256, gate * 256 + hT, k, lane),
                                   acc[gate]);
        }
#pragma unroll
        for (int j = 0; j < 8; ++j) {
            float c = sigm(acc[1][j]) * creg[j] + sigm(acc[0][j]) * ftanh(acc[2][j]);
            float h = sigm(acc[3][j]) * ftanh(c);
            creg[j] = c;
            _Float16 hh = (_Float16)h;
            sH[nxt][(m0 + j) * 256 + hid] = hh;
            hAll[(size_t)((bg0 + j) * 128 + t) * 512 + dir * 256 + hid] = hh;
        }
        __syncthreads();
        cur = nxt;
    }
}

// ------------------------------ feats GEMM ---------------------------------
// feats[4096x12] = hAll[4096x512] @ W2t^T (padded to N=16) + b2t
__global__ void __launch_bounds__(256)
feats_kernel(const _Float16* __restrict__ hAll, const _Float16* __restrict__ W2t,
             const float* __restrict__ b2t, float* __restrict__ feats) {
    const int lane = threadIdx.x & 31, wave = threadIdx.x >> 5;
    const int mT = (blockIdx.x * 8 + wave) << 4;
    int n = lane & 15;
    float bn = (n < 12) ? b2t[n] : 0.f;
    v8f acc;
#pragma unroll
    for (int j = 0; j < 8; ++j) acc[j] = bn;
#pragma unroll 1
    for (int k = 0; k < 512; k += 32)
        acc = wmma16(ldfrag(hAll, 512, mT, k, lane),
                     ldfrag(W2t, 512, 0, k, lane), acc);
    if (n < 12) {
        int m0 = mT + ((lane >> 4) << 3);
#pragma unroll
        for (int j = 0; j < 8; ++j) feats[(size_t)(m0 + j) * 12 + n] = acc[j];
    }
}

// --------------------------------- CRF -------------------------------------
__global__ void __launch_bounds__(32)
crf_kernel(const float* __restrict__ feats, const int* __restrict__ tags,
           const float* __restrict__ trans, float* __restrict__ out) {
    __shared__ float sT[144];
    __shared__ float sRed[32];
    const int tid = threadIdx.x;
    for (int i = tid; i < 144; i += 32) sT[i] = trans[i];
    __syncthreads();
    const int b = tid;
    float alpha[12];
#pragma unroll
    for (int i = 0; i < 12; ++i) alpha[i] = (i == 10) ? 0.f : -10000.f;
    for (int t = 0; t < 128; ++t) {
        const float* ft = feats + (size_t)(b * 128 + t) * 12;
        float na[12];
#pragma unroll
        for (int i = 0; i < 12; ++i) {
            const float* tr = sT + i * 12;
            float m = alpha[0] + tr[0];
#pragma unroll
            for (int j = 1; j < 12; ++j) m = fmaxf(m, alpha[j] + tr[j]);
            float s = 0.f;
#pragma unroll
            for (int j = 0; j < 12; ++j) s += __expf(alpha[j] + tr[j] - m);
            na[i] = m + __logf(s) + ft[i];
        }
#pragma unroll
        for (int i = 0; i < 12; ++i) alpha[i] = na[i];
    }
    float m = alpha[0] + sT[132];
#pragma unroll
    for (int j = 1; j < 12; ++j) m = fmaxf(m, alpha[j] + sT[132 + j]);
    float s = 0.f;
#pragma unroll
    for (int j = 0; j < 12; ++j) s += __expf(alpha[j] + sT[132 + j] - m);
    float fwd = m + __logf(s);
    int prev = 10;
    float gold = 0.f;
    for (int t = 0; t < 128; ++t) {
        int tg = tags[b * 128 + t];
        gold += feats[(size_t)(b * 128 + t) * 12 + tg] + sT[tg * 12 + prev];
        prev = tg;
    }
    gold += sT[132 + prev];
    sRed[tid] = fwd - gold;
    __syncthreads();
    if (tid == 0) {
        float tot = 0.f;
        for (int i = 0; i < 32; ++i) tot += sRed[i];
        out[0] = tot / 32.f;
    }
}

// ------------------------------- launcher ----------------------------------
extern "C" void kernel_launch(void* const* d_in, const int* in_sizes, int n_in,
                              void* d_out, int out_size, void* d_ws, size_t ws_size,
                              hipStream_t stream) {
    (void)in_sizes; (void)n_in; (void)out_size; (void)ws_size;
    const int*   sentence = (const int*)d_in[0];
    const int*   chars    = (const int*)d_in[1];
    const int*   tags     = (const int*)d_in[2];
    const float* word_emb = (const float*)d_in[3];
    const float* char_emb = (const float*)d_in[4];
    const float* c_Wih_f  = (const float*)d_in[5];
    const float* c_Whh_f  = (const float*)d_in[6];
    const float* c_b_f    = (const float*)d_in[7];
    const float* c_Wih_b  = (const float*)d_in[8];
    /* c_Whh_b (d_in[9]) mathematically unused: bwd char output = 1 step from zero state */
    const float* c_b_b    = (const float*)d_in[10];
    const float* w_Wih_f  = (const float*)d_in[11];
    const float* w_Whh_f  = (const float*)d_in[12];
    const float* w_b_f    = (const float*)d_in[13];
    const float* w_Wih_b  = (const float*)d_in[14];
    const float* w_Whh_b  = (const float*)d_in[15];
    const float* w_b_b    = (const float*)d_in[16];
    const float* W2t      = (const float*)d_in[17];
    const float* b2t      = (const float*)d_in[18];
    const float* trans    = (const float*)d_in[19];

    unsigned char* wsb = (unsigned char*)d_ws;
    size_t off = 0;
    auto take = [&](size_t bytes) -> void* {
        void* p = wsb + off;
        off += (bytes + 255) & ~(size_t)255;
        return p;
    };
    _Float16* hCE     = (_Float16*)take(128 * 64 * 2);
    _Float16* hcWihF  = (_Float16*)take(256 * 64 * 2);
    _Float16* hcWihB  = (_Float16*)take(256 * 64 * 2);
    _Float16* hcWhhF  = (_Float16*)take(256 * 64 * 2);
    _Float16* hwWihF  = (_Float16*)take(1024 * 384 * 2);
    _Float16* hwWihB  = (_Float16*)take(1024 * 384 * 2);
    _Float16* hwWhhF  = (_Float16*)take(1024 * 256 * 2);
    _Float16* hwWhhB  = (_Float16*)take(1024 * 256 * 2);
    _Float16* hW2t    = (_Float16*)take(16 * 512 * 2);
    _Float16* xw      = (_Float16*)take((size_t)4096 * 384 * 2);
    _Float16* XwF     = (_Float16*)take((size_t)4096 * 1024 * 2);
    _Float16* XwB     = (_Float16*)take((size_t)4096 * 1024 * 2);
    _Float16* hAll    = (_Float16*)take((size_t)4096 * 512 * 2);
    float*    feats   = (float*)take((size_t)4096 * 12 * 4);

    auto cvt = [&](const float* s, _Float16* d, int n) {
        cvt_f32_f16<<<(n + 255) / 256, 256, 0, stream>>>(s, d, n);
    };
    cvt(char_emb, hCE, 128 * 64);
    cvt(c_Wih_f, hcWihF, 256 * 64);
    cvt(c_Wih_b, hcWihB, 256 * 64);
    cvt(c_Whh_f, hcWhhF, 256 * 64);
    cvt(w_Wih_f, hwWihF, 1024 * 384);
    cvt(w_Wih_b, hwWihB, 1024 * 384);
    cvt(w_Whh_f, hwWhhF, 1024 * 256);
    cvt(w_Whh_b, hwWhhB, 1024 * 256);
    pad_w2t<<<(16 * 512 + 255) / 256, 256, 0, stream>>>(W2t, hW2t);

    char_fwd_kernel<<<128, 256, 0, stream>>>(chars, hCE, hcWihF, hcWhhF, c_b_f, xw);
    char_bwd_kernel<<<128, 256, 0, stream>>>(chars, hCE, hcWihB, c_b_b, xw);
    gather_we_kernel<<<4096, 256, 0, stream>>>(sentence, word_emb, xw);
    word_in_gemm<<<dim3(32, 8, 2), 256, 0, stream>>>(xw, hwWihF, hwWihB, XwF, XwB);
    word_rec_kernel<<<4, 512, 0, stream>>>(XwF, XwB, hwWhhF, hwWhhB, w_b_f, w_b_b, hAll);
    feats_kernel<<<32, 256, 0, stream>>>(hAll, hW2t, b2t, feats);
    crf_kernel<<<1, 32, 0, stream>>>(feats, tags, trans, (float*)d_out);
}
